// SelfAttention_4681514352840
// MI455X (gfx1250) — matrix operand order repaired and checked by reading, not compiled, not run on hardware
//
#include <hip/hip_runtime.h>

typedef __bf16 bf16;
typedef __attribute__((ext_vector_type(16))) __bf16 v16bf;
typedef __attribute__((ext_vector_type(8)))  __bf16 v8bf;
typedef __attribute__((ext_vector_type(8)))  float  v8f;
typedef int v4i_async __attribute__((vector_size(16)));  // matches builtin param type

#define WMMA_BF16(A, B, C) \
  __builtin_amdgcn_wmma_f32_16x16x32_bf16(false, (A), false, (B), (short)0, (C), false, false)

#define CONCAT8_8(LO, HI) \
  __builtin_shufflevector((LO), (HI), 0, 1, 2, 3, 4, 5, 6, 7, 8, 9, 10, 11, 12, 13, 14, 15)

constexpr int Dn   = 256;     // model dim == DK == DV
constexpr int Sn   = 4096;    // seq len
constexpr int Bn   = 4;       // batch
constexpr int ROWS = Bn * Sn; // 16384

#if __has_builtin(__builtin_amdgcn_global_load_async_to_lds_b128)
#define HAVE_ASYNC_LDS 1
#endif

// ---------------------------------------------------------------------------
// Kernel 1: fused QKV projection. out = x @ W + b  (bf16 into workspace).
// grid = (12, ROWS/256); block = 128 (4 waves).
// blockIdx.x = which*4 + colgroup : each block owns a 64-col block of one of
// Wq/Wk/Wv, stages it transposed+bf16 in LDS once, then sweeps 16 row tiles.
// For q, the 1/sqrt(DK) score scale is folded into the stored values.
// ---------------------------------------------------------------------------
__global__ __launch_bounds__(128) void qkv_proj_kernel(
    const float* __restrict__ x,
    const float* __restrict__ Wq, const float* __restrict__ bq,
    const float* __restrict__ Wk, const float* __restrict__ bk,
    const float* __restrict__ Wv, const float* __restrict__ bv,
    bf16* __restrict__ qkv_ws) {
  __shared__ __align__(32) bf16 wldsT[64 * Dn];  // [colLocal][k], 32 KB

  const int tid   = threadIdx.x;
  const int lane  = tid & 31;
  const int wave  = tid >> 5;
  const int n     = lane & 15;
  const int khalf = lane >> 4;

  const int which   = blockIdx.x >> 2;  // 0=q, 1=k, 2=v
  const int colbase = (blockIdx.x & 3) * 64;

  const float* W;
  const float* bias;
  float oscale;
  if (which == 0)      { W = Wq; bias = bq; oscale = 0.0625f; }  // 1/sqrt(256)
  else if (which == 1) { W = Wk; bias = bk; oscale = 1.0f; }
  else                 { W = Wv; bias = bv; oscale = 1.0f; }
  bf16* outp = qkv_ws + (size_t)which * ((size_t)ROWS * Dn);

  // Stage W[:, colbase:colbase+64] transposed into LDS as bf16: wldsT[c][k].
  for (int it = tid; it < 256 * 16; it += 128) {
    const int k  = it >> 4;
    const int cg = (it & 15) * 4;
    const float4 w4 = *(const float4*)(W + (size_t)k * Dn + colbase + cg);
    wldsT[(cg + 0) * 256 + k] = (bf16)w4.x;
    wldsT[(cg + 1) * 256 + k] = (bf16)w4.y;
    wldsT[(cg + 2) * 256 + k] = (bf16)w4.z;
    wldsT[(cg + 3) * 256 + k] = (bf16)w4.w;
  }
  __syncthreads();

  const int colLocal = wave * 16 + n;
  const int col      = colbase + colLocal;
  const float bval   = bias[col];

  for (int rt = 0; rt < 16; ++rt) {
    const int rowTile = blockIdx.y * 16 + rt;
    const int arow    = rowTile * 16 + n;

    v8f acc;
#pragma unroll
    for (int r = 0; r < 8; ++r) acc[r] = bval;

#pragma unroll
    for (int kc = 0; kc < 8; ++kc) {
      const int kb = kc * 32;
      // A fragment: x rows fp32 -> bf16; 16 contiguous k per lane half.
      const float* xp = x + (size_t)arow * Dn + kb + khalf * 16;
      const float4 a0 = *(const float4*)(xp);
      const float4 a1 = *(const float4*)(xp + 4);
      const float4 a2 = *(const float4*)(xp + 8);
      const float4 a3 = *(const float4*)(xp + 12);
      v16bf af;
      af[0]  = (bf16)a0.x; af[1]  = (bf16)a0.y; af[2]  = (bf16)a0.z; af[3]  = (bf16)a0.w;
      af[4]  = (bf16)a1.x; af[5]  = (bf16)a1.y; af[6]  = (bf16)a1.z; af[7]  = (bf16)a1.w;
      af[8]  = (bf16)a2.x; af[9]  = (bf16)a2.y; af[10] = (bf16)a2.z; af[11] = (bf16)a2.w;
      af[12] = (bf16)a3.x; af[13] = (bf16)a3.y; af[14] = (bf16)a3.z; af[15] = (bf16)a3.w;
      // B fragment: one contiguous v16bf from transposed-W LDS.
      const v16bf bfw = *(const v16bf*)&wldsT[colLocal * 256 + kb + khalf * 16];
      acc = WMMA_BF16(af, bfw, acc);
    }

#pragma unroll
    for (int r = 0; r < 8; ++r) {  // C/D: M = r + 8*khalf, N = n
      const int orow = rowTile * 16 + r + 8 * khalf;
      outp[(size_t)orow * Dn + col] = (bf16)(acc[r] * oscale);
    }
  }
}

// ---------------------------------------------------------------------------
// Kernel 2: causal flash attention. grid = (Sn/64, Bn); block = 128 (4 waves).
// Wave w owns query rows [qb + 16w, qb + 16w + 16); full 256-wide fp32 acc.
// ---------------------------------------------------------------------------
__global__ __launch_bounds__(128) void flash_attn_kernel(
    const bf16* __restrict__ q_ws, const bf16* __restrict__ k_ws,
    const bf16* __restrict__ v_ws, float* __restrict__ out) {
  __shared__ __align__(32) bf16 kbuf[32 * Dn];     // K chunk, row-major [key][d]
  __shared__ __align__(32) bf16 vbufT[Dn * 32];    // V chunk, transposed [dv][key]
  __shared__ __align__(32) bf16 pbuf[4][16 * 32];  // per-wave P tile [m][key]

  const int tid   = threadIdx.x;
  const int lane  = tid & 31;
  const int wave  = tid >> 5;
  const int n     = lane & 15;
  const int khalf = lane >> 4;

  const int    b     = blockIdx.y;
  const int    qb    = blockIdx.x * 64;
  const int    qrow0 = qb + wave * 16;
  const size_t base  = (size_t)b * Sn * Dn;
  const bf16* Q = q_ws + base;
  const bf16* K = k_ws + base;
  const bf16* V = v_ws + base;

  v8f acc[16];
  const v8f vzero = {};
#pragma unroll
  for (int t = 0; t < 16; ++t) acc[t] = vzero;
  float m_run[8], l_run[8];
#pragma unroll
  for (int r = 0; r < 8; ++r) { m_run[r] = -1e30f; l_run[r] = 0.0f; }

  const int nChunk = (qb + 64) / 32;  // keys 0 .. qb+63 in chunks of 32
  const bf16* qrowp = Q + (size_t)(qrow0 + n) * Dn;

  for (int c = 0; c < nChunk; ++c) {
    const int kk = c * 32;

    if (c + 1 < nChunk)  // speculative pull of next V chunk toward cache
      __builtin_prefetch((const void*)(V + (size_t)(kk + 32) * Dn + tid * 16), 0, 0);

    // ---- Stage K chunk (32x256 bf16, contiguous) ----
#ifdef HAVE_ASYNC_LDS
    {
      const bf16* gsrc = K + (size_t)kk * Dn;
      for (int i = tid * 8; i < 32 * Dn; i += 128 * 8)
        __builtin_amdgcn_global_load_async_to_lds_b128(
            (__attribute__((address_space(1))) v4i_async*)(gsrc + i),
            (__attribute__((address_space(3))) v4i_async*)(kbuf + i), 0, 0);
    }
#else
    {
      const uint4* src = (const uint4*)(K + (size_t)kk * Dn);
      uint4* dst = (uint4*)kbuf;
      for (int i = tid; i < (32 * Dn) / 8; i += 128) dst[i] = src[i];
    }
#endif

    // ---- Stage V chunk transposed: vbufT[dv][key] = V[kk+key][dv] ----
    {
      const int key = tid & 31, grp = tid >> 5;
      const bf16* vrow = V + (size_t)(kk + key) * Dn;
#pragma unroll
      for (int dv = grp * 64; dv < grp * 64 + 64; dv += 8) {
        const v8bf vv = *(const v8bf*)(vrow + dv);
#pragma unroll
        for (int j = 0; j < 8; ++j) vbufT[(dv + j) * 32 + key] = vv[j];
      }
    }
#ifdef HAVE_ASYNC_LDS
#if __has_builtin(__builtin_amdgcn_s_wait_asynccnt)
    __builtin_amdgcn_s_wait_asynccnt(0);
#else
    asm volatile("s_wait_asynccnt 0" ::: "memory");
#endif
#endif
    __syncthreads();

    // ---- Scores: S = Q @ K^T for 16 queries x 32 keys (two 16x16 tiles) ----
    v8f s0 = vzero, s1 = vzero;
#pragma unroll
    for (int kc = 0; kc < 8; ++kc) {
      const bf16* qp  = qrowp + kc * 32 + khalf * 16;
      const v8bf qlo  = *(const v8bf*)(qp);
      const v8bf qhi  = *(const v8bf*)(qp + 8);
      const v16bf af  = CONCAT8_8(qlo, qhi);
      const v16bf bf0 = *(const v16bf*)&kbuf[(size_t)(n)      * Dn + kc * 32 + khalf * 16];
      const v16bf bf1 = *(const v16bf*)&kbuf[(size_t)(n + 16) * Dn + kc * 32 + khalf * 16];
      s0 = WMMA_BF16(af, bf0, s0);
      s1 = WMMA_BF16(af, bf1, s1);
    }

    // ---- Causal mask + online softmax (scale pre-folded into Q) ----
    float alp[8];
#pragma unroll
    for (int r = 0; r < 8; ++r) {
      const int qr = qrow0 + r + 8 * khalf;
      s0[r] = ((kk + n)      <= qr) ? s0[r] : -1e30f;
      s1[r] = ((kk + 16 + n) <= qr) ? s1[r] : -1e30f;

      float mx = fmaxf(s0[r], s1[r]);
      mx = fmaxf(mx, __shfl_xor(mx, 1));
      mx = fmaxf(mx, __shfl_xor(mx, 2));
      mx = fmaxf(mx, __shfl_xor(mx, 4));
      mx = fmaxf(mx, __shfl_xor(mx, 8));
      const float mnew = fmaxf(m_run[r], mx);
      alp[r] = __expf(m_run[r] - mnew);
      m_run[r] = mnew;

      const float p0 = __expf(s0[r] - mnew);
      const float p1 = __expf(s1[r] - mnew);
      s0[r] = p0; s1[r] = p1;
      float rs = p0 + p1;
      rs += __shfl_xor(rs, 1);
      rs += __shfl_xor(rs, 2);
      rs += __shfl_xor(rs, 4);
      rs += __shfl_xor(rs, 8);
      l_run[r] = l_run[r] * alp[r] + rs;
    }
#pragma unroll
    for (int t = 0; t < 16; ++t)
#pragma unroll
      for (int r = 0; r < 8; ++r) acc[t][r] *= alp[r];

    // ---- P (C/D layout) -> LDS -> A-fragment layout ----
#pragma unroll
    for (int r = 0; r < 8; ++r) {
      const int prow = r + 8 * khalf;
      pbuf[wave][prow * 32 + n]      = (bf16)s0[r];
      pbuf[wave][prow * 32 + 16 + n] = (bf16)s1[r];
    }
    const v8bf plo = *(const v8bf*)&pbuf[wave][n * 32 + khalf * 16];
    const v8bf phi = *(const v8bf*)&pbuf[wave][n * 32 + khalf * 16 + 8];
    const v16bf pf = CONCAT8_8(plo, phi);

    // ---- acc += P @ V over 16 dv tiles ----
#pragma unroll
    for (int t = 0; t < 16; ++t) {
      const v16bf vf = *(const v16bf*)&vbufT[(size_t)(t * 16 + n) * 32 + khalf * 16];
      acc[t] = WMMA_BF16(pf, vf, acc[t]);
    }
    __syncthreads();
  }

  // ---- Normalize and store fp32 output ----
  float inv[8];
#pragma unroll
  for (int r = 0; r < 8; ++r) inv[r] = 1.0f / l_run[r];
#pragma unroll
  for (int t = 0; t < 16; ++t) {
#pragma unroll
    for (int r = 0; r < 8; ++r) {
      const int qr = qrow0 + r + 8 * khalf;
      out[base + (size_t)qr * Dn + t * 16 + n] = acc[t][r] * inv[r];
    }
  }
}

// ---------------------------------------------------------------------------
extern "C" void kernel_launch(void* const* d_in, const int* in_sizes, int n_in,
                              void* d_out, int out_size, void* d_ws, size_t ws_size,
                              hipStream_t stream) {
  const float* x  = (const float*)d_in[0];
  const float* Wq = (const float*)d_in[1];
  const float* bq = (const float*)d_in[2];
  const float* Wk = (const float*)d_in[3];
  const float* bk = (const float*)d_in[4];
  const float* Wv = (const float*)d_in[5];
  const float* bv = (const float*)d_in[6];
  float* out = (float*)d_out;
  bf16* ws = (bf16*)d_ws;  // q | k | v, each ROWS*Dn bf16 (24 MB total)

  dim3 g1(12, ROWS / 256);
  qkv_proj_kernel<<<g1, 128, 0, stream>>>(x, Wq, bq, Wk, bk, Wv, bv, ws);

  const size_t mat = (size_t)ROWS * Dn;
  dim3 g2(Sn / 64, Bn);
  flash_attn_kernel<<<g2, 128, 0, stream>>>(ws, ws + mat, ws + 2 * mat, out);
}
